// NTXentLoss_68504728371528
// MI455X (gfx1250) — matrix operand order repaired and checked by reading, not compiled, not run on hardware
//
#include <hip/hip_runtime.h>
#include <hip/hip_bf16.h>

typedef _Float16 v16h __attribute__((ext_vector_type(16)));
typedef _Float16 v8h  __attribute__((ext_vector_type(8)));
typedef float    v8f  __attribute__((ext_vector_type(8)));
typedef int      v4i  __attribute__((ext_vector_type(4)));

#define DIM 128
#define TEMP_INV 2.0f      // 1 / 0.5
#define EPS_NRM 1e-8f
#define CCHUNKS 8          // column chunks (grid.y) for occupancy
#define SC 64              // columns staged in LDS per stage (4 WMMA tiles)

#if __has_builtin(__builtin_amdgcn_global_load_async_to_lds_b128) && \
    __has_builtin(__builtin_amdgcn_s_wait_asynccnt)
#define USE_ASYNC_LDS 1
#else
#define USE_ASYNC_LDS 0
#endif

union F16x16 { v16h v; v8h h[2]; };

#if USE_ASYNC_LDS
__device__ __forceinline__ void async_copy_b128(const _Float16* g, _Float16* l) {
    typedef __attribute__((address_space(1))) v4i* gptr_t;
    typedef __attribute__((address_space(3))) v4i* lptr_t;
    __builtin_amdgcn_global_load_async_to_lds_b128((gptr_t)g, (lptr_t)l, 0, 0);
}
#endif

// exp(sim / T) folded into a single mul + v_exp_f32 (exp2 based)
__device__ __forceinline__ float exp_simT(float c) {
#if __has_builtin(__builtin_amdgcn_exp2f)
    return __builtin_amdgcn_exp2f(c * 2.8853900817779268f);  // (1/T)*log2(e)
#else
    return __expf(c * TEMP_INV);
#endif
}

__device__ __forceinline__ const float* rep_row(const float* zis, const float* zjs,
                                                int r, int Bsz) {
    // rep = concat(zjs, zis)
    return (r < Bsz) ? (zjs + (size_t)r * DIM) : (zis + (size_t)(r - Bsz) * DIM);
}

// ---------------- Kernel 1: row norms + f16 normalized copy --------------------
__global__ void ntx_normalize(const float* __restrict__ zis,
                              const float* __restrict__ zjs,
                              _Float16* __restrict__ nf16,
                              float* __restrict__ inv_norm,
                              int Bsz) {
    const int wave = threadIdx.x >> 5;
    const int lane = threadIdx.x & 31;
    const int row  = blockIdx.x * 8 + wave;
    const float* src = rep_row(zis, zjs, row, Bsz);

    float4 x = *(const float4*)(src + lane * 4);
    float s = x.x * x.x + x.y * x.y + x.z * x.z + x.w * x.w;
    #pragma unroll
    for (int m = 16; m >= 1; m >>= 1) s += __shfl_xor(s, m, 32);

    const float inv = 1.0f / fmaxf(sqrtf(s), EPS_NRM);
    if (lane == 0) inv_norm[row] = inv;

    struct H4 { _Float16 a, b, c, d; } h;
    h.a = (_Float16)(x.x * inv);
    h.b = (_Float16)(x.y * inv);
    h.c = (_Float16)(x.z * inv);
    h.d = (_Float16)(x.w * inv);
    *(H4*)(nf16 + (size_t)row * DIM + lane * 4) = h;
}

// ---------------- Kernel 2: positives pos[i]/T in fp32 -------------------------
__global__ void ntx_positives(const float* __restrict__ zis,
                              const float* __restrict__ zjs,
                              const float* __restrict__ inv_norm,
                              float* __restrict__ posT,
                              int Bsz) {
    const int wave = threadIdx.x >> 5;
    const int lane = threadIdx.x & 31;
    const int row  = blockIdx.x * 8 + wave;
    const int prt  = (row + Bsz) & (2 * Bsz - 1);

    const float* a = rep_row(zis, zjs, row, Bsz);
    const float* b = rep_row(zis, zjs, prt, Bsz);
    float4 xa = *(const float4*)(a + lane * 4);
    float4 xb = *(const float4*)(b + lane * 4);
    float s = xa.x * xb.x + xa.y * xb.y + xa.z * xb.z + xa.w * xb.w;
    #pragma unroll
    for (int m = 16; m >= 1; m >>= 1) s += __shfl_xor(s, m, 32);

    if (lane == 0)
        posT[row] = s * inv_norm[row] * inv_norm[prt] * TEMP_INV;
}

// ---------------- Kernel 3: fused WMMA GEMM + masked exp row-sums --------------
// grid = (n2/128, CCHUNKS), block = 256 (8 waves). Wave owns 16 rows; block
// covers a 1024-column chunk, staged through double-buffered LDS (async copy).
__global__ void ntx_simsum(const _Float16* __restrict__ nf16,
                           float* __restrict__ part,
                           int Bsz) {
    const int n2      = 2 * Bsz;
    const int tid     = threadIdx.x;
    const int lane    = tid & 31;
    const int wave    = tid >> 5;
    const int l16     = lane & 15;
    const bool hi     = lane >= 16;

    // wave-uniform row base as an SGPR -> scalar branches for masking
    const int rowBase = __builtin_amdgcn_readfirstlane((blockIdx.x * 8 + wave) * 16);

    const int chunkCols = n2 / CCHUNKS;               // 1024
    const int chunkBase = blockIdx.y * chunkCols;
    const int nStages   = chunkCols / SC;             // 16

    // masked diagonals live in exactly these two global column tiles (SGPRs)
    const int sSelf = rowBase >> 4;
    const int sPos  = ((rowBase + Bsz) & (n2 - 1)) >> 4;

    __shared__ _Float16 sbuf[2][SC * DIM];            // 2 x 16 KB

    // ---- A fragments (16-bit A 16x32, contiguous k = 16*(lane/16) + s), loop-invariant in VGPRs
    const _Float16* arow = nf16 + (size_t)(rowBase + l16) * DIM;
    const int ab = hi ? 16 : 0;
    F16x16 afrag[4];
    #pragma unroll
    for (int k = 0; k < 4; ++k) {
        const int ko = 32 * k;
        afrag[k].h[0] = *(const v8h*)(arow + ko + ab);
        afrag[k].h[1] = *(const v8h*)(arow + ko + 8 + ab);
    }

    // ---- cooperative stage loader: SC rows x 256B, 4 x b128 per thread
    auto stage_issue = [&](int buf, int s) {
        const int c0 = chunkBase + s * SC;
        #pragma unroll
        for (int j = 0; j < 4; ++j) {
            const int idx = tid + 256 * j;            // 0..1023 b128 chunks
            const int row = idx >> 4;
            const int off = (idx & 15) * 8;           // halves
            const _Float16* g = nf16 + (size_t)(c0 + row) * DIM + off;
            _Float16* l = &sbuf[buf][row * DIM + off];
#if USE_ASYNC_LDS
            async_copy_b128(g, l);
#else
            *(v8h*)l = *(const v8h*)g;
#endif
        }
    };

    float rowAcc[8];
    #pragma unroll
    for (int i = 0; i < 8; ++i) rowAcc[i] = 0.0f;

    const int bb = hi ? 16 : 0;

    stage_issue(0, 0);
    if (nStages > 1) stage_issue(1, 1);

    for (int s = 0; s < nStages; ++s) {
#if USE_ASYNC_LDS
        // our own async loads: leave only the next stage's 4 outstanding
        if (s + 1 < nStages) __builtin_amdgcn_s_wait_asynccnt(4);
        else                 __builtin_amdgcn_s_wait_asynccnt(0);
#endif
        __syncthreads();                              // stage s visible to all

        const _Float16* sb = sbuf[s & 1];
        const int tileBase = (chunkBase + s * SC) >> 4;

        #pragma unroll
        for (int tt = 0; tt < 4; ++tt) {
            const int gt = tileBase + tt;             // scalar tile index
            const _Float16* brow = sb + ((tt * 16 + l16) << 7) + bb;

            // software-pipelined K: two alternating fragment buffers so each
            // s_wait_dscnt only covers loads issued two WMMAs earlier.
            F16x16 b0, b1;
            b0.h[0] = *(const v8h*)(brow +   0);      // k0
            b0.h[1] = *(const v8h*)(brow +   8);
            b1.h[0] = *(const v8h*)(brow +  32);      // k1
            b1.h[1] = *(const v8h*)(brow +  40);

            v8f c = {};
            c = __builtin_amdgcn_wmma_f32_16x16x32_f16(
                    false, afrag[0].v, false, b0.v, (short)0, c, false, false);
            b0.h[0] = *(const v8h*)(brow +  64);      // k2
            b0.h[1] = *(const v8h*)(brow +  72);
            c = __builtin_amdgcn_wmma_f32_16x16x32_f16(
                    false, afrag[1].v, false, b1.v, (short)0, c, false, false);
            b1.h[0] = *(const v8h*)(brow +  96);      // k3
            b1.h[1] = *(const v8h*)(brow + 104);
            c = __builtin_amdgcn_wmma_f32_16x16x32_f16(
                    false, afrag[2].v, false, b0.v, (short)0, c, false, false);
            c = __builtin_amdgcn_wmma_f32_16x16x32_f16(
                    false, afrag[3].v, false, b1.v, (short)0, c, false, false);

            const int n = (gt << 4) + l16;
            if (gt == sSelf || gt == sPos) {          // scalar branch, rare
                #pragma unroll
                for (int i = 0; i < 8; ++i) {
                    const int r = rowBase + i + (hi ? 8 : 0);
                    const bool masked =
                        (n == r) | (n == ((r + Bsz) & (n2 - 1)));
                    rowAcc[i] += masked ? 0.0f : exp_simT(c[i]);
                }
            } else {                                  // steady state: no cmps
                #pragma unroll
                for (int i = 0; i < 8; ++i)
                    rowAcc[i] += exp_simT(c[i]);
            }
        }

        __syncthreads();                              // done reading sbuf[s&1]
        if (s + 2 < nStages) stage_issue(s & 1, s + 2);
    }

    // reduce over the 16 lanes of each C-matrix half
    #pragma unroll
    for (int i = 0; i < 8; ++i) {
        float v = rowAcc[i];
        #pragma unroll
        for (int m = 1; m < 16; m <<= 1) v += __shfl_xor(v, m, 32);
        rowAcc[i] = v;
    }

    if (l16 == 0) {
        const int r0 = rowBase + (hi ? 8 : 0);
        float* dst = part + (size_t)blockIdx.y * n2;
        #pragma unroll
        for (int i = 0; i < 8; ++i) dst[r0 + i] = rowAcc[i];
    }
}

// ---------------- Kernel 4: deterministic finalize + reduction -----------------
__global__ void ntx_finalize(const float* __restrict__ part,
                             const float* __restrict__ posT,
                             float* __restrict__ out, int n2) {
    __shared__ float sm[256];
    float s = 0.0f;
    for (int r = threadIdx.x; r < n2; r += 256) {
        float tot = 0.0f;
        #pragma unroll
        for (int c = 0; c < CCHUNKS; ++c) tot += part[(size_t)c * n2 + r];
        const float pT = posT[r];
        s += logf(tot + __expf(pT)) - pT;             // lse - pos/T
    }
    sm[threadIdx.x] = s;
    __syncthreads();
    for (int off = 128; off > 0; off >>= 1) {
        if (threadIdx.x < off) sm[threadIdx.x] += sm[threadIdx.x + off];
        __syncthreads();
    }
    if (threadIdx.x == 0) out[0] = sm[0] / (float)n2;
}

extern "C" void kernel_launch(void* const* d_in, const int* in_sizes, int n_in,
                              void* d_out, int out_size, void* d_ws, size_t ws_size,
                              hipStream_t stream) {
    const float* zis = (const float*)d_in[0];
    const float* zjs = (const float*)d_in[1];
    // d_in[2] mask / d_in[3] labels are structural; reproduced analytically.
    const int Bsz = in_sizes[0] / DIM;      // 4096
    const int n2  = 2 * Bsz;                // 8192

    char* ws = (char*)d_ws;
    _Float16* nf16  = (_Float16*)ws;                        // n2*DIM*2  (2 MB)
    float* inv_norm = (float*)(ws + (size_t)n2 * DIM * 2);  // n2
    float* posT     = inv_norm + n2;                        // n2
    float* part     = posT + n2;                            // CCHUNKS*n2 (256 KB)
    float* out      = (float*)d_out;

    ntx_normalize<<<n2 / 8, 256, 0, stream>>>(zis, zjs, nf16, inv_norm, Bsz);
    ntx_positives<<<n2 / 8, 256, 0, stream>>>(zis, zjs, inv_norm, posT, Bsz);
    dim3 g3(n2 / 128, CCHUNKS);
    ntx_simsum  <<<g3,     256, 0, stream>>>(nf16, part, Bsz);
    ntx_finalize<<<1,      256, 0, stream>>>(part, posT, out, n2);
}
